// STSampleNet_27805618274763
// MI455X (gfx1250) — matrix operand order repaired and checked by reading, not compiled, not run on hardware
//
#include <hip/hip_runtime.h>
#include <cstdint>

#define B_   64
#define T_   9
#define R_   374
#define F_   128
#define K_   299
#define BT_  (B_*T_)        // 576
#define NROW (B_*T_*R_)     // 215424
#define EPS_ 1.17549435e-38f

typedef __attribute__((ext_vector_type(16))) _Float16 v16h;
typedef __attribute__((ext_vector_type(8)))  float    v8f;
typedef unsigned int u32x4 __attribute__((ext_vector_type(4)));
typedef int          i32x4 __attribute__((ext_vector_type(4)));
typedef int          i32x8 __attribute__((ext_vector_type(8)));

union Frag { v16h h; uint32_t u[8]; };

__device__ __forceinline__ float gelu_exact(float x) {
    return 0.5f * x * (1.0f + erff(x * 0.7071067811865476f));
}

// ---------------------------------------------------------------------------
// Tensor Data Mover: 1-D tile copy of `nelem` 2-byte elements, global -> LDS.
// D# per CDNA5 ISA ch.8: group0 = {count|flags, lds_addr, global_addr[31:0],
// global_addr[56:32] | type=2<<30}; group1 packs data_size=2B, tensor_dim0 =
// tile_dim0 = nelem (1-D), dim1 = 1. Issued by one wave; EXEC ignored;
// completion tracked via TENSORcnt.
// ---------------------------------------------------------------------------
#if __has_builtin(__builtin_amdgcn_tensor_load_to_lds)
#define HAVE_TDM 1
__device__ __forceinline__ void tdm_copy_2b(const void* gsrc, void* ldst,
                                            uint32_t nelem /* <= 65535 */)
{
    const uint64_t ga = (uint64_t)(uintptr_t)gsrc;
    const uint32_t la = (uint32_t)(uintptr_t)ldst;   // generic LDS ptr: addr[31:0] = LDS offset

    u32x4 g0;
    g0[0] = 1u;                                           // count=1, user mode, no gather
    g0[1] = la;                                           // lds_addr  [63:32]
    g0[2] = (uint32_t)ga;                                 // global_addr[95:64] = addr[31:0]
    g0[3] = (uint32_t)((ga >> 32) & 0x01FFFFFFu)          // addr[56:32] in bits[24:0]
          | (2u << 30);                                   // type=2 ("image") bits[127:126]

    i32x8 g1;
    g1[0] = (int)(1u << 16);                              // workgroup_mask=0, data_size=1 (2B)
    g1[1] = (int)((nelem & 0xFFFFu) << 16);               // tensor_dim0[15:0]  -> bits[63:48]
    g1[2] = (int)(((nelem >> 16) & 0xFFFFu) | (1u << 16));// tensor_dim0[31:16], tensor_dim1=1
    g1[3] = (int)(nelem << 16);                           // tile_dim0 -> bits[127:112]
    g1[4] = 1;                                            // tile_dim1=1, tile_dim2 unused
    g1[5] = (int)nelem;                                   // tensor_dim0_stride[31:0]
    g1[6] = 0;                                            // stride hi / dim1_stride lo
    g1[7] = 0;

    i32x4 gz = {};
#if __clang_major__ >= 23
    i32x8 gz8 = {};
    __builtin_amdgcn_tensor_load_to_lds(g0, g1, gz, gz, gz8, 0);
#else
    __builtin_amdgcn_tensor_load_to_lds(g0, g1, gz, gz, 0);
#endif
}
#endif

__device__ __forceinline__ void wait_tensor0() {
#if __has_builtin(__builtin_amdgcn_s_wait_tensorcnt)
    __builtin_amdgcn_s_wait_tensorcnt(0);
#else
    asm volatile("s_wait_tensorcnt 0x0" ::: "memory");
#endif
}

// ---------------------------------------------------------------------------
// kPrep: w1 (128x128 f32 [k][n]) -> w1T f16 [n][k];  w2 (128x32) -> w2T f16
// ---------------------------------------------------------------------------
__global__ void kPrep(const float* __restrict__ w1, const float* __restrict__ w2,
                      _Float16* __restrict__ w1T, _Float16* __restrict__ w2T)
{
    const int t = threadIdx.x;
    for (int i = t; i < 128 * 128; i += 256) {
        int k = i >> 7, n = i & 127;
        w1T[n * 128 + k] = (_Float16)w1[i];
    }
    for (int i = t; i < 128 * 32; i += 256) {
        int k = i >> 5, n = i & 31;
        w2T[n * 128 + k] = (_Float16)w2[i];
    }
}

// ---------------------------------------------------------------------------
// kA: LayerNorm + GEMM1(128x128, f16 WMMA) + GELU -> h1 (N x 128, f16)
//     one block = 16 rows, 256 threads = 8 waves; w1T staged to LDS by TDM
// ---------------------------------------------------------------------------
__global__ __launch_bounds__(256) void kA(const float* __restrict__ x,
                                          const float* __restrict__ lnw,
                                          const float* __restrict__ lnb,
                                          const float* __restrict__ b1,
                                          const _Float16* __restrict__ w1T,
                                          _Float16* __restrict__ h1)
{
    __shared__ _Float16 Alds[16][128];   // 4 KB
    __shared__ _Float16 Blds[128][128];  // 32 KB (w1T, n-major)
    __shared__ float redS[256], redQ[256];
    __shared__ float mu[16], rs[16];

    const int t    = threadIdx.x;
    const int row0 = blockIdx.x * 16;

    // kick off w1T -> LDS DMA from wave 0 (TDM; overlaps with LayerNorm below)
#ifdef HAVE_TDM
    if (t < 32) tdm_copy_2b(w1T, &Blds[0][0], 128 * 128);
#else
    {
        const uint4* src = (const uint4*)w1T;
        uint4*       dst = (uint4*)&Blds[0][0];
        #pragma unroll
        for (int i = 0; i < 8; ++i) dst[t + 256 * i] = src[t + 256 * i];
    }
#endif

    // LayerNorm: 16 threads per row, 8 features each
    const int r = t >> 4, c = t & 15;
    const int row = row0 + r;
    const int f0  = c * 8;
    float4 a0 = *(const float4*)(x + (size_t)row * F_ + f0);
    float4 a1 = *(const float4*)(x + (size_t)row * F_ + f0 + 4);
    {
        float s = a0.x + a0.y + a0.z + a0.w + a1.x + a1.y + a1.z + a1.w;
        float q = a0.x*a0.x + a0.y*a0.y + a0.z*a0.z + a0.w*a0.w
                + a1.x*a1.x + a1.y*a1.y + a1.z*a1.z + a1.w*a1.w;
        redS[t] = s; redQ[t] = q;
    }
    __syncthreads();
    if (c == 0) {
        float ts = 0.f, tq = 0.f;
        #pragma unroll
        for (int i = 0; i < 16; ++i) { ts += redS[(r << 4) + i]; tq += redQ[(r << 4) + i]; }
        float m = ts * (1.0f / 128.0f);
        float v = tq * (1.0f / 128.0f) - m * m;
        mu[r] = m;
        rs[r] = rsqrtf(v + 1e-5f);
    }
    __syncthreads();
    {
        float m = mu[r], rsi = rs[r];
        float xv[8] = {a0.x, a0.y, a0.z, a0.w, a1.x, a1.y, a1.z, a1.w};
        #pragma unroll
        for (int i = 0; i < 8; ++i) {
            int f = f0 + i;
            float y = (xv[i] - m) * rsi * lnw[f] + lnb[f];
            Alds[r][f] = (_Float16)y;
        }
    }
#ifdef HAVE_TDM
    if (t < 32) wait_tensor0();          // wave 0 drains TENSORcnt before barrier
#endif
    __syncthreads();

    // WMMA: D(16x16) = A(16x128) x B(128x16); preload all 4 k-step fragments
    const int wave = t >> 5, lane = t & 31;
    const int n0 = wave * 16;
    const int lanelo = lane & 15;
    const int hiA = (lane >= 16) ? 8  : 0;
    const int hiB = (lane >= 16) ? 16 : 0;

    Frag afr[4], bfr[4];
    #pragma unroll
    for (int s4 = 0; s4 < 4; ++s4) {
        const int kk = s4 * 32;
        #pragma unroll
        for (int v = 0; v < 8; ++v) {
            int ka = kk + 2 * v + ((v >= 4) ? 8 : 0) + hiA;   // A: lane=M, k-pairs
            afr[s4].u[v] = *(const uint32_t*)&Alds[lanelo][ka];
            int kb = kk + 2 * v + ((v >= 4) ? 8 : 0) + hiA;   // B: lane=N, k-pairs
            bfr[s4].u[v] = *(const uint32_t*)&Blds[n0 + lanelo][kb];
        }
    }
    v8f acc = {};
    #pragma unroll
    for (int s4 = 0; s4 < 4; ++s4)
        acc = __builtin_amdgcn_wmma_f32_16x16x32_f16(false, afr[s4].h, false, bfr[s4].h,
                                                     (short)0, acc, false, false);

    const int n  = n0 + lanelo;
    const float bn = b1[n];
    const int mbase = (lane >= 16) ? 8 : 0;
    #pragma unroll
    for (int v = 0; v < 8; ++v) {
        int orow = row0 + mbase + v;                           // C/D: M = v + 8*(lane>=16)
        float val = gelu_exact(acc[v] + bn);
        h1[(size_t)orow * F_ + n] = (_Float16)val;
    }
}

// ---------------------------------------------------------------------------
// kG: glob[bt][j] = mean over R of h1[bt,:,64+j]   (deterministic fixed-order)
// ---------------------------------------------------------------------------
__global__ __launch_bounds__(256) void kG(const _Float16* __restrict__ h1,
                                          _Float16* __restrict__ glob)
{
    __shared__ float red[256];
    const int bt = blockIdx.x;
    const int t = threadIdx.x;
    const int col = t & 63, grp = t >> 6;
    float s = 0.f;
    for (int r = grp; r < R_; r += 4)
        s += (float)h1[((size_t)bt * R_ + r) * F_ + 64 + col];
    red[t] = s;
    __syncthreads();
    if (t < 64) {
        float tot = red[t] + red[t + 64] + red[t + 128] + red[t + 192];
        glob[bt * 64 + t] = (_Float16)(tot * (1.0f / (float)R_));
    }
}

// ---------------------------------------------------------------------------
// kB: [local | glob] -> GEMM2(128x32 WMMA)+GELU -> GEMM3(32x2)+softmax+gumbel
//     one block = 16 rows, 64 threads = 2 waves; w2T staged to LDS by TDM
// ---------------------------------------------------------------------------
__global__ __launch_bounds__(64) void kB(const _Float16* __restrict__ h1,
                                         const _Float16* __restrict__ glob,
                                         const _Float16* __restrict__ w2T,
                                         const float* __restrict__ b2,
                                         const float* __restrict__ w3,
                                         const float* __restrict__ b3,
                                         const float* __restrict__ u,
                                         float* __restrict__ s0)
{
    __shared__ _Float16 Alds[16][128];   // 4 KB
    __shared__ _Float16 Blds[32][128];   // 8 KB (w2T n-major)
    __shared__ float    h2[16][32];      // 2 KB
    const int t    = threadIdx.x;
    const int row0 = blockIdx.x * 16;

#ifdef HAVE_TDM
    if (t < 32) tdm_copy_2b(w2T, &Blds[0][0], 32 * 128);
#else
    {
        const uint4* src = (const uint4*)w2T;
        uint4*       dst = (uint4*)&Blds[0][0];
        #pragma unroll
        for (int i = 0; i < 8; ++i) dst[t + 64 * i] = src[t + 64 * i];
    }
#endif

    // assemble A2 = [h1 local half | glob broadcast]: 4 threads/row, 32 halfs each
    {
        const int r = t >> 2, seg = t & 3;
        const int row = row0 + r;
        const int bt  = row / R_;
        const uint4* src = (seg < 2)
            ? (const uint4*)(h1 + (size_t)row * F_ + seg * 32)
            : (const uint4*)(glob + bt * 64 + (seg - 2) * 32);
        uint4* dst = (uint4*)&Alds[r][seg * 32];
        dst[0] = src[0]; dst[1] = src[1]; dst[2] = src[2]; dst[3] = src[3];
    }
#ifdef HAVE_TDM
    if (t < 32) wait_tensor0();
#endif
    __syncthreads();

    const int wave = t >> 5, lane = t & 31;
    const int n0 = wave * 16;
    const int lanelo = lane & 15;
    const int hiA = (lane >= 16) ? 8  : 0;
    const int hiB = (lane >= 16) ? 16 : 0;

    Frag afr[4], bfr[4];
    #pragma unroll
    for (int s4 = 0; s4 < 4; ++s4) {
        const int kk = s4 * 32;
        #pragma unroll
        for (int v = 0; v < 8; ++v) {
            int ka = kk + 2 * v + ((v >= 4) ? 8 : 0) + hiA;
            afr[s4].u[v] = *(const uint32_t*)&Alds[lanelo][ka];
            int kb = kk + 2 * v + ((v >= 4) ? 8 : 0) + hiA;
            bfr[s4].u[v] = *(const uint32_t*)&Blds[n0 + lanelo][kb];
        }
    }
    v8f acc = {};
    #pragma unroll
    for (int s4 = 0; s4 < 4; ++s4)
        acc = __builtin_amdgcn_wmma_f32_16x16x32_f16(false, afr[s4].h, false, bfr[s4].h,
                                                     (short)0, acc, false, false);

    const int n = n0 + lanelo;
    const float bn = b2[n];
    const int mbase = (lane >= 16) ? 8 : 0;
    #pragma unroll
    for (int v = 0; v < 8; ++v)
        h2[mbase + v][n] = gelu_exact(acc[v] + bn);
    __syncthreads();

    // GEMM3 (32 -> 2), softmax -> p1, gumbel
    if (t < 16) {
        const int row = row0 + t;
        float l0 = b3[0], l1 = b3[1];
        #pragma unroll
        for (int g = 0; g < 32; ++g) {
            float hv = h2[t][g];
            l0 += hv * w3[2 * g];
            l1 += hv * w3[2 * g + 1];
        }
        float sc  = 1.0f / (1.0f + expf(l0 - l1));   // softmax p[...,1]
        float gum = -logf(-logf(u[row]));
        s0[row] = sc + gum;
    }
}

// ---------------------------------------------------------------------------
// kC: subset operator (299 softmax iterations over R=374) + exact top-k(299)
//     one wave32 per (b,t); 12 element slots per lane; shuffle reductions
// ---------------------------------------------------------------------------
__device__ __forceinline__ float wmaxf(float v) {
    #pragma unroll
    for (int o = 16; o > 0; o >>= 1) v = fmaxf(v, __shfl_xor(v, o, 32));
    return v;
}
__device__ __forceinline__ float wsumf(float v) {
    #pragma unroll
    for (int o = 16; o > 0; o >>= 1) v += __shfl_xor(v, o, 32);
    return v;
}

__global__ __launch_bounds__(32) void kC(const float* __restrict__ s0,
                                         float* __restrict__ prob,
                                         int* __restrict__ idx)
{
    __shared__ float vals[384];
    const int bt   = blockIdx.x;
    const int lane = threadIdx.x;

    float s[12], khot[12], oneh[12];
    bool valid[12];
    #pragma unroll
    for (int j = 0; j < 12; ++j) {
        int i = lane + 32 * j;
        valid[j] = (i < R_);
        s[j]    = valid[j] ? s0[bt * R_ + i] : -1.0e30f;
        khot[j] = 0.f;
        oneh[j] = 0.f;
    }

    for (int it = 0; it < K_; ++it) {
        float lm = -1.0e30f;
        #pragma unroll
        for (int j = 0; j < 12; ++j) {
            s[j] += logf(fmaxf(1.0f - oneh[j], EPS_));
            lm = fmaxf(lm, s[j]);
        }
        float m = wmaxf(lm);
        float e[12];
        float ls = 0.f;
        #pragma unroll
        for (int j = 0; j < 12; ++j) {
            e[j] = valid[j] ? expf(s[j] - m) : 0.f;
            ls += e[j];
        }
        float inv = 1.0f / wsumf(ls);
        #pragma unroll
        for (int j = 0; j < 12; ++j) {
            oneh[j] = e[j] * inv;
            khot[j] += oneh[j];
        }
    }

    #pragma unroll
    for (int j = 0; j < 12; ++j) {
        int i = lane + 32 * j;
        vals[i] = valid[j] ? khot[j] : -1.0e30f;
        if (valid[j]) prob[bt * R_ + i] = khot[j];
    }
    __syncthreads();

    // rank-based top-k: rank = #{greater} + #{equal with smaller index}
    #pragma unroll
    for (int j = 0; j < 12; ++j) {
        int i = lane + 32 * j;
        if (!valid[j]) continue;
        float v = vals[i];
        int rank = 0;
        for (int q = 0; q < R_; ++q) {
            float vq = vals[q];
            rank += (vq > v) || (vq == v && q < i);
        }
        if (rank < K_) idx[bt * K_ + rank] = i;
    }
}

// ---------------------------------------------------------------------------
extern "C" void kernel_launch(void* const* d_in, const int* in_sizes, int n_in,
                              void* d_out, int out_size, void* d_ws, size_t ws_size,
                              hipStream_t stream)
{
    const float* x   = (const float*)d_in[0];
    const float* u   = (const float*)d_in[1];
    const float* lnw = (const float*)d_in[2];
    const float* lnb = (const float*)d_in[3];
    const float* w1  = (const float*)d_in[4];
    const float* b1  = (const float*)d_in[5];
    const float* w2  = (const float*)d_in[6];
    const float* b2  = (const float*)d_in[7];
    const float* w3  = (const float*)d_in[8];
    const float* b3  = (const float*)d_in[9];

    char* ws = (char*)d_ws;
    _Float16* w1T  = (_Float16*)(ws);                      // 32 KB
    _Float16* w2T  = (_Float16*)(ws + 32768);              // 8 KB
    _Float16* glob = (_Float16*)(ws + 40960);              // 72 KB (576*64 f16)
    _Float16* h1   = (_Float16*)(ws + 114688);             // 55.1 MB (N*128 f16)
    float*    s0   = (float*)(ws + 55263232ull);           // 0.86 MB (N f32)

    float* prob = (float*)d_out;                           // N floats
    int*   idx  = (int*)(prob + NROW);                     // BT*K int32

    hipLaunchKernelGGL(kPrep, dim3(1),         dim3(256), 0, stream, w1, w2, w1T, w2T);
    hipLaunchKernelGGL(kA,    dim3(NROW / 16), dim3(256), 0, stream, x, lnw, lnb, b1, w1T, h1);
    hipLaunchKernelGGL(kG,    dim3(BT_),       dim3(256), 0, stream, h1, glob);
    hipLaunchKernelGGL(kB,    dim3(NROW / 16), dim3(64),  0, stream, h1, glob, w2T, b2, w3, b3, u, s0);
    hipLaunchKernelGGL(kC,    dim3(BT_),       dim3(32),  0, stream, s0, prob, idx);
}
